// StyleBlock_68487548502383
// MI455X (gfx1250) — compile-verified, matrix operand order checked by reading, not run on hardware
//
#include <hip/hip_runtime.h>
#include <stdint.h>

typedef __attribute__((ext_vector_type(4)))  _Float16 v4h;
typedef __attribute__((ext_vector_type(8)))  _Float16 v8h;
typedef __attribute__((ext_vector_type(16))) _Float16 v16h;
typedef __attribute__((ext_vector_type(8)))  float    v8f;
typedef __attribute__((ext_vector_type(4)))  int      v4i;

#define B_ 8
#define H_ 64
#define W_ 64
#define C_ 512            // Cin == Cout == wdim
#define INV_SQRT_WDIM 0.044194173824159216f   // 1/sqrt(512)
#define SQRT2         1.4142135623730951f
#define WSCALE        0.014731391274719735f   // 1/sqrt(9*512)
#define WSCALE2       (1.0f / 4608.0f)
#define NSTEPS 144                            // 9 taps * (512/32) k-steps

// ---- CDNA5 async global->LDS path (guarded; fallback = register staging) ----
#if defined(__has_builtin)
#if __has_builtin(__builtin_amdgcn_global_load_async_to_lds_b128) && \
    __has_builtin(__builtin_amdgcn_s_wait_asynccnt)
#define USE_ASYNC_LDS 1
#endif
#endif

#ifdef USE_ASYNC_LDS
__device__ __forceinline__ void async_cp_b128(const void* g, void* l) {
  // per-lane 16B DMA: LDS[l] = MEM[g], tracked by ASYNCcnt
  __builtin_amdgcn_global_load_async_to_lds_b128(
      (__attribute__((address_space(1))) v4i*)(uintptr_t)g,
      (__attribute__((address_space(3))) v4i*)(unsigned int)(uintptr_t)l,
      0, 0);
}
#endif

// ---------------------------------------------------------------------------
// s[b,c] = leaky(w[b,:] @ sw[:,c]/sqrt(512) + sb[c], 0.2) * sqrt(2)
// ---------------------------------------------------------------------------
__global__ void style_kernel(const float* __restrict__ w,
                             const float* __restrict__ sw,
                             const float* __restrict__ sb,
                             float* __restrict__ s) {
  int b = blockIdx.x;
  int c = threadIdx.x;
  const float* wr = w + b * C_;
  float acc = 0.0f;
  for (int k = 0; k < C_; ++k) acc = fmaf(wr[k], sw[k * C_ + c], acc);
  float v = acc * INV_SQRT_WDIM + sb[c];
  v = (v > 0.0f) ? v : 0.2f * v;
  s[b * C_ + c] = v * SQRT2;
}

// ---------------------------------------------------------------------------
// d[b,o] = rsqrt( (1/4608) * sum_{tap,i} cw[tap,i,o]^2 * s[b,i]^2 + 1e-8 )
// ---------------------------------------------------------------------------
__global__ void demod_kernel(const float* __restrict__ cw,
                             const float* __restrict__ s,
                             float* __restrict__ d) {
  int b = blockIdx.x;
  int o = threadIdx.x;
  const float* srow = s + b * C_;
  float acc = 0.0f;
  for (int i = 0; i < C_; ++i) {
    float si = srow[i];
    float s2 = si * si;
#pragma unroll
    for (int t = 0; t < 9; ++t) {
      float wv = cw[((size_t)(t * C_ + i)) * C_ + o];
      acc = fmaf(wv * wv, s2, acc);
    }
  }
  d[b * C_ + o] = rsqrtf(acc * WSCALE2 + 1e-8f);
}

// ---------------------------------------------------------------------------
// xm[b,h,w,c] = half( x[b,h,w,c] * s[b,c] )   (NHWC, vectorized x4)
// ---------------------------------------------------------------------------
__global__ void modulate_kernel(const float4* __restrict__ x4,
                                const float* __restrict__ s,
                                v4h* __restrict__ xm4) {
  size_t i = (size_t)blockIdx.x * blockDim.x + threadIdx.x;  // 2^22 quads
  size_t total = ((size_t)B_ * H_ * W_ * C_) >> 2;
  if (i >= total) return;
  int c4 = (int)((i & 127) << 2);          // channel quad
  int b  = (int)(i >> 19);                 // H*W*C/4 = 2^19 per image
  float4 v = x4[i];
  const float* sr = s + b * C_ + c4;
  v4h o;
  o[0] = (_Float16)(v.x * sr[0]);
  o[1] = (_Float16)(v.y * sr[1]);
  o[2] = (_Float16)(v.z * sr[2]);
  o[3] = (_Float16)(v.w * sr[3]);
  xm4[i] = o;
}

// ---------------------------------------------------------------------------
// wkT[tap][o][i] = half( cw[tap][i][o] / sqrt(4608) )  (i-contiguous rows)
// ---------------------------------------------------------------------------
__global__ void wprep_kernel(const float* __restrict__ cw,
                             _Float16* __restrict__ wkT) {
  size_t i = (size_t)blockIdx.x * blockDim.x + threadIdx.x;
  size_t total = (size_t)9 * C_ * C_;
  if (i >= total) return;
  int tap = (int)(i / (C_ * C_));
  int r   = (int)(i - (size_t)tap * C_ * C_);
  int o   = r >> 9;
  int ii  = r & (C_ - 1);
  wkT[i] = (_Float16)(cw[((size_t)(tap * C_ + ii)) * C_ + o] * WSCALE);
}

// ---------------------------------------------------------------------------
// Implicit-GEMM 3x3 conv + fused demod/noise/bias/leaky tail.
// Block = 256 threads (8 waves) -> macro-tile 128 pixels x 128 outputs.
// Wave grid 4(M) x 2(N): wave tile 32(M) x 64(N) = 2x4 WMMA f32 accumulators.
// K pipeline: 144 steps of 32; async global->LDS DMA (or reg staging),
// double-buffered LDS, one s_barrier per step, copies overlap the 8 WMMAs.
// ---------------------------------------------------------------------------
__device__ __forceinline__ v16h load_frag(const _Float16* row, int hi) {
  // ISA 16-bit fragment layout: lane holds K in [8*hi, 8*hi+8) then
  // [16+8*hi, 16+8*hi+8) as 16 packed f16.
  v8h l0 = *(const v8h*)(row + (hi << 3));
  v8h l1 = *(const v8h*)(row + 16 + (hi << 3));
  return __builtin_shufflevector(l0, l1, 0, 1, 2, 3, 4, 5, 6, 7,
                                 8, 9, 10, 11, 12, 13, 14, 15);
}

#ifdef USE_ASYNC_LDS
__device__ __forceinline__ void stage_async(int s, int b, int h0, int bn,
                                            int sp, int sk,
                                            const _Float16* __restrict__ xm,
                                            const _Float16* __restrict__ wkT,
                                            _Float16* bufA, _Float16* bufB) {
  int tap = s >> 4;
  int k0  = (s & 15) << 5;
  int dh  = tap / 3 - 1;
  int dw  = tap % 3 - 1;
  int hh  = h0 + (sp >> 6) + dh;
  int ww  = (sp & 63) + dw;
  _Float16* la = bufA + (sp << 5) + sk;
  if (hh >= 0 && hh < H_ && ww >= 0 && ww < W_) {
    const _Float16* ga = xm + ((((size_t)b * H_ + hh) * W_ + ww) << 9)
                         + k0 + sk;
    async_cp_b128(ga, la);
    async_cp_b128(ga + 8, la + 8);
  } else {                       // halo: DMA can't zero-fill, do it with DS
    *(v8h*)la = (v8h){};
    *(v8h*)(la + 8) = (v8h){};
  }
  const _Float16* gb = wkT + (size_t)tap * C_ * C_
                       + ((size_t)(bn + sp) << 9) + k0 + sk;
  _Float16* lb = bufB + (sp << 5) + sk;
  async_cp_b128(gb, lb);
  async_cp_b128(gb + 8, lb + 8);
}
#else
__device__ __forceinline__ void stage_load(int s, int b, int h0, int bn,
                                           int sp, int sk,
                                           const _Float16* __restrict__ xm,
                                           const _Float16* __restrict__ wkT,
                                           v8h& a0, v8h& a1,
                                           v8h& b0, v8h& b1) {
  int tap = s >> 4;
  int k0  = (s & 15) << 5;
  int dh  = tap / 3 - 1;
  int dw  = tap % 3 - 1;
  int hh  = h0 + (sp >> 6) + dh;
  int ww  = (sp & 63) + dw;
  a0 = (v8h){};
  a1 = (v8h){};
  if (hh >= 0 && hh < H_ && ww >= 0 && ww < W_) {
    const v8h* pa = (const v8h*)(xm + ((((size_t)b * H_ + hh) * W_ + ww) << 9)
                                 + k0 + sk);
    a0 = pa[0];
    a1 = pa[1];
  }
  const v8h* pb = (const v8h*)(wkT + (size_t)tap * C_ * C_
                               + ((size_t)(bn + sp) << 9) + k0 + sk);
  b0 = pb[0];
  b1 = pb[1];
}
#endif

__launch_bounds__(256)
__global__ void conv_wmma_kernel(const _Float16* __restrict__ xm,
                                 const _Float16* __restrict__ wkT,
                                 const float* __restrict__ dS,
                                 const float* __restrict__ noise,
                                 const float* __restrict__ sn,
                                 const float* __restrict__ bias,
                                 float* __restrict__ out) {
  __shared__ _Float16 lA[2 * 128 * 32];   // double-buffered A  16 KB
  __shared__ _Float16 lB[2 * 128 * 32];   // double-buffered B  16 KB

  int bm = blockIdx.x >> 2;               // 256 M-tiles of 128 pixels
  int bn = (blockIdx.x & 3) << 7;         // 4 N-tiles of 128 channels
  int P0 = bm << 7;                       // first flat pixel of tile
  int b  = P0 >> 12;                      // 4096 pixels per image
  int h0 = (P0 & 4095) >> 6;              // tile = rows h0, h0+1 (full width)

  int tid  = threadIdx.x;
  int lane = tid & 31;
  int wv   = tid >> 5;
  int m0   = (wv >> 1) << 5;              // wave's 32-pixel sub-tile
  int nw   = (wv & 1) << 6;               // wave's 64-channel sub-tile
  int hi   = lane >> 4;
  int lo16 = lane & 15;

  int sp = tid >> 1;                      // staging row 0..127 (pixel / n)
  int sk = (tid & 1) << 4;                // staging k-half 0/16

  v8f acc[2][4] = {};

#ifdef USE_ASYNC_LDS
  stage_async(0, b, h0, bn, sp, sk, xm, wkT, lA, lB);
#else
  v8h a0, a1, b0, b1;
  stage_load(0, b, h0, bn, sp, sk, xm, wkT, a0, a1, b0, b1);
#endif

#pragma unroll 2
  for (int s = 0; s < NSTEPS; ++s) {
    _Float16* bufA = lA + ((s & 1) << 12);
    _Float16* bufB = lB + ((s & 1) << 12);

#ifdef USE_ASYNC_LDS
    // drain this step's DMA, then sync so every wave's tile is visible
    __builtin_amdgcn_s_wait_asynccnt(0);
    __syncthreads();
    if (s + 1 < NSTEPS) {
      stage_async(s + 1, b, h0, bn, sp, sk, xm, wkT,
                  lA + (((s + 1) & 1) << 12), lB + (((s + 1) & 1) << 12));
    }
#else
    {
      v8h* da = (v8h*)(bufA + (sp << 5) + sk);
      da[0] = a0;
      da[1] = a1;
      v8h* db = (v8h*)(bufB + (sp << 5) + sk);
      db[0] = b0;
      db[1] = b1;
    }
    __syncthreads();
    if (s + 1 < NSTEPS) {
      stage_load(s + 1, b, h0, bn, sp, sk, xm, wkT, a0, a1, b0, b1);
    }
#endif

    // fragments + 8 WMMAs
    v16h bF[4];
#pragma unroll
    for (int j = 0; j < 4; ++j) {
      bF[j] = load_frag(bufB + ((nw + (j << 4) + lo16) << 5), hi);
    }
#pragma unroll
    for (int t = 0; t < 2; ++t) {
      v16h aF = load_frag(bufA + ((m0 + (t << 4) + lo16) << 5), hi);
#pragma unroll
      for (int j = 0; j < 4; ++j) {
        acc[t][j] = __builtin_amdgcn_wmma_f32_16x16x32_f16(
            false, aF, false, bF[j], (short)0, acc[t][j], false, false);
      }
    }
  }

  // ---- fused epilogue: y*d + sn*noise + bias, leaky(0.2) ----
  float dv[4], sv[4], bv[4];
  int nn[4];
#pragma unroll
  for (int j = 0; j < 4; ++j) {
    int n = bn + nw + (j << 4) + lo16;
    nn[j] = n;
    dv[j] = dS[b * C_ + n];
    sv[j] = sn[n];
    bv[j] = bias[n];
  }
#pragma unroll
  for (int t = 0; t < 2; ++t) {
#pragma unroll
    for (int r = 0; r < 8; ++r) {
      int P = P0 + m0 + (t << 4) + (hi << 3) + r;  // flat pixel == noise index
      float nz = noise[P];
      size_t obase = ((size_t)P) << 9;
#pragma unroll
      for (int j = 0; j < 4; ++j) {
        float y = acc[t][j][r] * dv[j] + sv[j] * nz + bv[j];
        out[obase + nn[j]] = (y > 0.0f) ? y : 0.2f * y;
      }
    }
  }
}

// ---------------------------------------------------------------------------
extern "C" void kernel_launch(void* const* d_in, const int* in_sizes, int n_in,
                              void* d_out, int out_size, void* d_ws,
                              size_t ws_size, hipStream_t stream) {
  const float* x     = (const float*)d_in[0];  // [8,64,64,512]
  const float* w     = (const float*)d_in[1];  // [8,512]
  const float* noise = (const float*)d_in[2];  // [8,64,64,1]
  const float* sw    = (const float*)d_in[3];  // [512,512]
  const float* sb    = (const float*)d_in[4];  // [512]
  const float* cw    = (const float*)d_in[5];  // [3,3,512,512]
  const float* sn    = (const float*)d_in[6];  // [1,1,1,512]
  const float* bias  = (const float*)d_in[7];  // [512]
  float* out = (float*)d_out;

  // workspace layout
  float*    sPtr  = (float*)d_ws;                           // 4096 f32
  float*    dPtr  = sPtr + B_ * C_;                         // 4096 f32
  _Float16* xmPtr = (_Float16*)(dPtr + B_ * C_);            // 16.78M f16
  _Float16* wkPtr = xmPtr + (size_t)B_ * H_ * W_ * C_;      // 2.36M f16

  style_kernel<<<B_, C_, 0, stream>>>(w, sw, sb, sPtr);
  demod_kernel<<<B_, C_, 0, stream>>>(cw, sPtr, dPtr);

  size_t nxm4 = ((size_t)B_ * H_ * W_ * C_) >> 2;
  modulate_kernel<<<(int)(nxm4 / 256), 256, 0, stream>>>(
      (const float4*)x, sPtr, (v4h*)xmPtr);

  size_t nwk = (size_t)9 * C_ * C_;
  wprep_kernel<<<(int)((nwk + 255) / 256), 256, 0, stream>>>(cw, wkPtr);

  conv_wmma_kernel<<<(B_ * H_ * W_ / 128) * (C_ / 128), 256, 0, stream>>>(
      xmPtr, wkPtr, dPtr, noise, sn, bias, out);
}
